// TokenMergeModule_53034256171280
// MI455X (gfx1250) — compile-verified, matrix operand order checked by reading, not run on hardware
//
#include <hip/hip_runtime.h>
#include <hip/hip_bf16.h>
#include <math.h>

// Problem constants (fixed by the reference).
#define BB   4
#define SSZ  4096
#define DDIM 512
#define GDIM 64
#define RSEL 1024
#define NPAIR (3 * SSZ - 6)       // 12282
#define PADP  16384               // pairs padded to pow2 for bitonic sort
#define SOUT  (SSZ - RSEL)        // 3072

typedef __attribute__((ext_vector_type(16))) _Float16 v16h;
typedef __attribute__((ext_vector_type(8)))  float    v8f;
typedef __attribute__((ext_vector_type(4)))  float    v4f;
typedef __attribute__((ext_vector_type(4)))  unsigned int v4u;
typedef __attribute__((ext_vector_type(8)))  int      v8i;
typedef __attribute__((ext_vector_type(4)))  int      v4i;

#if defined(__HIP_DEVICE_COMPILE__) && __has_builtin(__builtin_amdgcn_tensor_load_to_lds) && __has_builtin(__builtin_amdgcn_s_wait_tensorcnt)
#define K1_TDM 1
#endif

// ---------------------------------------------------------------------------
// K1: g = x @ W^T via WMMA f32_16x16x32_f16.  M = B*S = 16384, N = G = 64,
// K = D = 512.  One block = one 16-row M-tile; its 4 waves each own one of
// the 4 N-tiles.  The 16x512 f32 A-slab (32 KB) is staged into LDS once per
// block via the Tensor Data Mover (TENSORcnt + workgroup barrier), then A
// fragments are read from LDS (ds_load_b128) and B (W rows) from global
// (128 KB, L2/WGP$-resident).
// ---------------------------------------------------------------------------
__global__ __launch_bounds__(128) void k_gemm_g(const float* __restrict__ x,
                                                const float* __restrict__ W,
                                                float* __restrict__ g) {
    __shared__ float atile[16 * DDIM];              // 32 KB, LDS offset 0
    const int wid   = threadIdx.x >> 5;             // 0..3 -> N-tile
    const int lane  = threadIdx.x & 31;
    const int mBase = blockIdx.x * 16;              // 1024 M-tiles
    const int nBase = wid * 16;
    const int half  = lane >> 4;
    const int l15   = lane & 15;

#ifdef K1_TDM
    // Keep the LDS view typed addrspace(3) so A reads lower to ds_load, and
    // escape the pointer through an asm memory clobber so the TDM's (invisible
    // to the compiler) write to atile can't be folded to undef.
    typedef __attribute__((address_space(3))) const v4f* ldsv4_t;
    ldsv4_t av = (ldsv4_t)atile;
    if (wid == 0) {
        // D# per CDNA5 ISA ch.8: 2D tensor, data_size=4B, row length 512,
        // tile = 512 x 16 rows, contiguous (stride 512), dest LDS offset 0.
        const unsigned long long ga =
            (unsigned long long)(const void*)(x + (size_t)mBase * DDIM);
        v4u g0;
        g0[0] = 1u;                                          // count=1
        g0[1] = 0u;                                          // lds_addr = 0
        g0[2] = (unsigned)(ga & 0xFFFFFFFFu);                // global_addr[31:0]
        g0[3] = (unsigned)((ga >> 32) & 0x01FFFFFFu) | (2u << 30); // ga[56:32] | type=2
        v8i g1;
        g1[0] = 0x00020000;          // wg_mask=0, data_size=2 (4B)
        g1[1] = (int)(512u << 16);   // tensor_dim0 = 512 (bits 79:48 low part)
        g1[2] = (int)(16384u << 16); // tensor_dim1 = 16384 rows (bits 111:80)
        g1[3] = (int)(512u << 16);   // tile_dim0 = 512 (bits 127:112)
        g1[4] = 16;                  // tile_dim1 = 16 rows
        g1[5] = 512;                 // tensor_dim0_stride = 512
        g1[6] = 0;
        g1[7] = 0;
        v4i z4 = {0, 0, 0, 0};
#if __clang_major__ >= 23
        v8i z8 = {0, 0, 0, 0, 0, 0, 0, 0};
        __builtin_amdgcn_tensor_load_to_lds(g0, g1, z4, z4, z8, 0);
#else
        __builtin_amdgcn_tensor_load_to_lds(g0, g1, z4, z4, 0);
#endif
    }
    __builtin_amdgcn_s_wait_tensorcnt(0);
    asm volatile("" : "+v"(av) :: "memory");
    __syncthreads();
#else
    // Fallback: cooperative float4-wide copy into LDS.
    {
        const v4f* srcp = (const v4f*)(x + (size_t)mBase * DDIM);
        for (int t = threadIdx.x; t < 16 * DDIM / 4; t += 128)
            ((v4f*)atile)[t] = srcp[t];
    }
    const v4f* av = (const v4f*)atile;
    __syncthreads();
#endif

    // A fragment rows come from LDS; B fragment rows (W) from global.
    const int arow  = l15 * (DDIM / 4);             // v4f index of A row
    const float* brow = W + (size_t)(nBase + l15) * DDIM;

    v8f acc = {};
#pragma unroll 4
    for (int k0 = 0; k0 < DDIM; k0 += 32) {
        const int abase = arow + (k0 >> 2) + half * 2;
        v4f av0 = av[abase + 0];
        v4f av1 = av[abase + 1];
        v4f av2 = av[abase + 4];                    // +16 floats
        v4f av3 = av[abase + 5];
        const float4* bp0 = (const float4*)(brow + k0 + half * 8);
        const float4* bp1 = (const float4*)(brow + k0 + 16 + half * 8);
        float4 bv0 = bp0[0], bv1 = bp0[1], bv2 = bp1[0], bv3 = bp1[1];

        v16h a, b;
        a[0]  = (_Float16)av0.x; a[1]  = (_Float16)av0.y; a[2]  = (_Float16)av0.z; a[3]  = (_Float16)av0.w;
        a[4]  = (_Float16)av1.x; a[5]  = (_Float16)av1.y; a[6]  = (_Float16)av1.z; a[7]  = (_Float16)av1.w;
        a[8]  = (_Float16)av2.x; a[9]  = (_Float16)av2.y; a[10] = (_Float16)av2.z; a[11] = (_Float16)av2.w;
        a[12] = (_Float16)av3.x; a[13] = (_Float16)av3.y; a[14] = (_Float16)av3.z; a[15] = (_Float16)av3.w;
        b[0]  = (_Float16)bv0.x; b[1]  = (_Float16)bv0.y; b[2]  = (_Float16)bv0.z; b[3]  = (_Float16)bv0.w;
        b[4]  = (_Float16)bv1.x; b[5]  = (_Float16)bv1.y; b[6]  = (_Float16)bv1.z; b[7]  = (_Float16)bv1.w;
        b[8]  = (_Float16)bv2.x; b[9]  = (_Float16)bv2.y; b[10] = (_Float16)bv2.z; b[11] = (_Float16)bv2.w;
        b[12] = (_Float16)bv3.x; b[13] = (_Float16)bv3.y; b[14] = (_Float16)bv3.z; b[15] = (_Float16)bv3.w;

        acc = __builtin_amdgcn_wmma_f32_16x16x32_f16(
            /*neg_a=*/false, a, /*neg_b=*/false, b,
            /*c_mod=*/(short)0, acc, /*reuse_a=*/false, /*reuse_b=*/false);
    }

    // C/D layout: VGPR r holds M = mBase + half*8 + r at column nBase + l15.
    const int gcol = nBase + l15;
    const int mrow = mBase + half * 8;
#pragma unroll
    for (int r = 0; r < 8; ++r)
        g[(size_t)(mrow + r) * GDIM + gcol] = acc[r];
}

// ---------------------------------------------------------------------------
// K2: per-row norm + normalize in place.  One wave per row (G=64 -> 2/lane).
// ---------------------------------------------------------------------------
__global__ __launch_bounds__(256) void k_norm(float* __restrict__ g,
                                              float* __restrict__ gnorm) {
    const int row  = blockIdx.x * 8 + (threadIdx.x >> 5);
    const int lane = threadIdx.x & 31;
    float v0 = g[(size_t)row * GDIM + lane];
    float v1 = g[(size_t)row * GDIM + 32 + lane];
    float ss = v0 * v0 + v1 * v1;
#pragma unroll
    for (int off = 16; off > 0; off >>= 1) ss += __shfl_xor(ss, off, 32);
    float n = sqrtf(ss);
    if (lane == 0) gnorm[row] = n;
    float inv = 1.0f / fmaxf(n, 1e-12f);
    g[(size_t)row * GDIM + lane]      = v0 * inv;
    g[(size_t)row * GDIM + 32 + lane] = v1 * inv;
}

// ---------------------------------------------------------------------------
// K3: pair similarities + 64-bit sort keys. One wave per (b, padded-pair).
// key = orderable(-sim) << 32 | (i*S+j)  (ascending sort => desc sim, lex tie)
// ---------------------------------------------------------------------------
__global__ __launch_bounds__(256) void k_sim(const float* __restrict__ gn,
                                             unsigned long long* __restrict__ keys,
                                             unsigned* __restrict__ vals) {
    const int e    = blockIdx.x * 8 + (threadIdx.x >> 5);   // b*PADP + p
    const int lane = threadIdx.x & 31;
    const int b = e >> 14;
    const int p = e & (PADP - 1);
    if (p >= NPAIR) {
        if (lane == 0) { keys[e] = ~0ull; vals[e] = 0xFFFFFFFFu; }
        return;
    }
    int i, j;
    if (p < SSZ - 1)          { i = p;                 j = i + 1; }
    else if (p < 2 * SSZ - 3) { i = p - (SSZ - 1);     j = i + 2; }
    else                      { i = p - (2 * SSZ - 3); j = i + 3; }

    const float* gi = gn + ((size_t)b * SSZ + i) * GDIM;
    const float* gj = gn + ((size_t)b * SSZ + j) * GDIM;
    float s = gi[lane] * gj[lane] + gi[lane + 32] * gj[lane + 32];
#pragma unroll
    for (int off = 16; off > 0; off >>= 1) s += __shfl_xor(s, off, 32);
    if (lane == 0) {
        unsigned u = __float_as_uint(-s);
        u = (u & 0x80000000u) ? ~u : (u | 0x80000000u);
        keys[e] = ((unsigned long long)u << 32) | (unsigned)(i * SSZ + j);
        vals[e] = (unsigned)p;
    }
}

// ---------------------------------------------------------------------------
// K4: one bitonic compare-exchange pass over all batches.
// ---------------------------------------------------------------------------
__global__ __launch_bounds__(256) void k_bitonic(unsigned long long* __restrict__ keys,
                                                 unsigned* __restrict__ vals,
                                                 int kstep, int jstep) {
    const int t = blockIdx.x * 256 + threadIdx.x;   // B*PADP threads
    const int b = t >> 14;
    const int i = t & (PADP - 1);
    const int ixj = i ^ jstep;
    if (ixj <= i) return;
    const size_t base = (size_t)b * PADP;
    unsigned long long ki = keys[base + i], kj = keys[base + ixj];
    const bool up = (i & kstep) == 0;
    if ((ki > kj) == up) {
        keys[base + i] = kj; keys[base + ixj] = ki;
        unsigned vi = vals[base + i], vj = vals[base + ixj];
        vals[base + i] = vj; vals[base + ixj] = vi;
    }
}

__global__ void k_init_partner(int* __restrict__ partner) {
    const int t = blockIdx.x * 256 + threadIdx.x;
    if (t < BB * SSZ) partner[t] = -1;
}

// ---------------------------------------------------------------------------
// K5: greedy selection (serial per batch; 4096-bit used mask in registers).
// partner[s] = j  if s merged with j (s keeps, weights from gnorm)
// partner[s] = -2 if s removed (was a sel_j);  -1 untouched.
// ---------------------------------------------------------------------------
__global__ void k_select(const unsigned* __restrict__ vals,
                         int* __restrict__ partner,
                         int* __restrict__ sel_i, int* __restrict__ sel_j) {
    const int b = threadIdx.x;
    if (b >= BB) return;
    unsigned long long used[SSZ / 64];
    for (int q = 0; q < SSZ / 64; ++q) used[q] = 0ull;
    int cnt = 0;
    const unsigned* v = vals + (size_t)b * PADP;
    for (int p = 0; p < PADP && cnt < RSEL; ++p) {
        unsigned pe = v[p];
        if (pe >= (unsigned)NPAIR) continue;
        int i, j;
        if (pe < SSZ - 1)               { i = (int)pe;                 j = i + 1; }
        else if (pe < 2u * SSZ - 3u)    { i = (int)pe - (SSZ - 1);     j = i + 2; }
        else                            { i = (int)pe - (2 * SSZ - 3); j = i + 3; }
        bool ui = (used[i >> 6] >> (i & 63)) & 1ull;
        bool uj = (used[j >> 6] >> (j & 63)) & 1ull;
        if (!ui && !uj) {
            used[i >> 6] |= 1ull << (i & 63);
            used[j >> 6] |= 1ull << (j & 63);
            sel_i[b * RSEL + cnt] = i;
            sel_j[b * RSEL + cnt] = j;
            partner[b * SSZ + i] = j;
            partner[b * SSZ + j] = -2;
            ++cnt;
        }
    }
}

// ---------------------------------------------------------------------------
// K6: stable compaction of kept tokens via LDS block scan (per batch).
// ---------------------------------------------------------------------------
__global__ __launch_bounds__(1024) void k_compact(const int* __restrict__ partner,
                                                  int* __restrict__ keep_idx) {
    __shared__ int lsum[1024];
    const int b = blockIdx.x, t = threadIdx.x;
    const int base = b * SSZ + t * 4;
    int k[4], s = 0;
#pragma unroll
    for (int q = 0; q < 4; ++q) { k[q] = (partner[base + q] != -2); s += k[q]; }
    lsum[t] = s;
    __syncthreads();
    for (int off = 1; off < 1024; off <<= 1) {
        int vv = (t >= off) ? lsum[t - off] : 0;
        __syncthreads();
        lsum[t] += vv;
        __syncthreads();
    }
    int rank = lsum[t] - s;
#pragma unroll
    for (int q = 0; q < 4; ++q)
        if (k[q]) { if (rank < SOUT) keep_idx[b * SOUT + rank] = t * 4 + q; ++rank; }
}

// ---------------------------------------------------------------------------
// K7: gather + weighted merge of x rows. One block per output row, B128 I/O.
// ---------------------------------------------------------------------------
__global__ __launch_bounds__(128) void k_gather_x(const float* __restrict__ x,
                                                  const float* __restrict__ gnorm,
                                                  const int* __restrict__ keep_idx,
                                                  const int* __restrict__ partner,
                                                  float* __restrict__ xout) {
    const int o = blockIdx.x;             // B*SOUT rows
    const int b = o / SOUT;
    const int row = keep_idx[o];
    const int pj  = partner[b * SSZ + row];
    const v4f* xi = (const v4f*)(x + ((size_t)b * SSZ + row) * DDIM);
    v4f vi = xi[threadIdx.x];
    v4f out;
    if (pj >= 0) {
        const float wi = gnorm[b * SSZ + row];
        const float wj = gnorm[b * SSZ + pj];
        const v4f* xj = (const v4f*)(x + ((size_t)b * SSZ + pj) * DDIM);
        v4f vj = xj[threadIdx.x];
        const float inv = 1.0f / (wi + wj + 1e-8f);
        out = (vi * wi + vj * wj) * inv;
    } else {
        out = vi;
    }
    ((v4f*)(xout + (size_t)o * DDIM))[threadIdx.x] = out;
}

// ---------------------------------------------------------------------------
// K8: gather + add of source rows (bandwidth-dominant: ~0.45 GB total).
// source (256 MB) exceeds the 192 MB L2 -> stream with non-temporal hints.
// ---------------------------------------------------------------------------
__global__ __launch_bounds__(256) void k_gather_src(const float* __restrict__ src,
                                                    const int* __restrict__ keep_idx,
                                                    const int* __restrict__ partner,
                                                    float* __restrict__ sout) {
    const int o = blockIdx.x;             // B*SOUT rows of length S
    const int b = o / SOUT;
    const int row = keep_idx[o];
    const int pj  = partner[b * SSZ + row];
    const v4f* si = (const v4f*)(src + ((size_t)b * SSZ + row) * SSZ);
    v4f* dp = (v4f*)(sout + (size_t)o * SSZ);
    if (pj >= 0) {
        const v4f* sj = (const v4f*)(src + ((size_t)b * SSZ + pj) * SSZ);
#pragma unroll
        for (int it = 0; it < 4; ++it) {
            const int idx = it * 256 + threadIdx.x;
            if (it < 3) {
                __builtin_prefetch(&si[idx + 256], 0, 0);
                __builtin_prefetch(&sj[idx + 256], 0, 0);
            }
            v4f a = __builtin_nontemporal_load(si + idx);
            v4f c = __builtin_nontemporal_load(sj + idx);
            v4f r = a + c;
            __builtin_nontemporal_store(r, dp + idx);
        }
    } else {
#pragma unroll
        for (int it = 0; it < 4; ++it) {
            const int idx = it * 256 + threadIdx.x;
            if (it < 3) __builtin_prefetch(&si[idx + 256], 0, 0);
            v4f a = __builtin_nontemporal_load(si + idx);
            __builtin_nontemporal_store(a, dp + idx);
        }
    }
}

// K9: position id gather.
__global__ void k_pos(const int* __restrict__ pos_in,
                      const int* __restrict__ keep_idx,
                      int* __restrict__ pos_out) {
    const int t = blockIdx.x * 256 + threadIdx.x;
    if (t >= BB * SOUT) return;
    const int b = t / SOUT;
    pos_out[t] = pos_in[b * SSZ + keep_idx[t]];
}

// ---------------------------------------------------------------------------
extern "C" void kernel_launch(void* const* d_in, const int* in_sizes, int n_in,
                              void* d_out, int out_size, void* d_ws, size_t ws_size,
                              hipStream_t stream) {
    (void)in_sizes; (void)n_in; (void)out_size; (void)ws_size;
    const float* x   = (const float*)d_in[0];
    const float* src = (const float*)d_in[1];
    const int*   pos = (const int*)  d_in[2];
    const float* W   = (const float*)d_in[3];

    char* ws = (char*)d_ws;
    float*              g        = (float*)(ws + 0);                    // 4 MiB
    float*              gnorm    = (float*)(ws + 4194304);              // 64 KiB
    unsigned long long* keys     = (unsigned long long*)(ws + 4259840); // 512 KiB
    unsigned*           vals     = (unsigned*)(ws + 4784128);           // 256 KiB
    int*                sel_i    = (int*)(ws + 5046272);                // 16 KiB
    int*                sel_j    = (int*)(ws + 5062656);                // 16 KiB
    int*                partner  = (int*)(ws + 5079040);                // 64 KiB
    int*                keep_idx = (int*)(ws + 5144576);                // 48 KiB

    float* x_out = (float*)d_out;
    float* s_out = x_out + (size_t)BB * SOUT * DDIM;
    int*   p_out = (int*)(s_out + (size_t)BB * SOUT * SSZ);

    k_gemm_g<<<1024, 128, 0, stream>>>(x, W, g);
    k_norm<<<2048, 256, 0, stream>>>(g, gnorm);
    k_sim<<<8192, 256, 0, stream>>>(g, keys, vals);
    for (int k = 2; k <= PADP; k <<= 1)
        for (int j = k >> 1; j > 0; j >>= 1)
            k_bitonic<<<256, 256, 0, stream>>>(keys, vals, k, j);
    k_init_partner<<<64, 256, 0, stream>>>(partner);
    k_select<<<1, 32, 0, stream>>>(vals, partner, sel_i, sel_j);
    k_compact<<<BB, 1024, 0, stream>>>(partner, keep_idx);
    k_gather_x<<<BB * SOUT, 128, 0, stream>>>(x, gnorm, keep_idx, partner, x_out);
    k_gather_src<<<BB * SOUT, 256, 0, stream>>>(src, keep_idx, partner, s_out);
    k_pos<<<48, 256, 0, stream>>>(pos, keep_idx, p_out);
}
